// BertSelfAttention_6433861009571
// MI455X (gfx1250) — matrix operand order repaired and checked by reading, not compiled, not run on hardware
//
#include <hip/hip_runtime.h>

// ---- problem constants (BERT-base self-attention, B=4, S=2048) ----
#define HID   768
#define NHEAD 12
#define HDIM  64
#define SEQ   2048
#define BATCH 4
#define NROW  (BATCH * SEQ)   // 8192

typedef _Float16 v4h  __attribute__((ext_vector_type(4)));
typedef _Float16 v8h  __attribute__((ext_vector_type(8)));
typedef _Float16 v16h __attribute__((ext_vector_type(16)));
typedef float    v8f  __attribute__((ext_vector_type(8)));

static __device__ __forceinline__ v16h cat16(v8h lo, v8h hi) {
  return __builtin_shufflevector(lo, hi, 0,1,2,3,4,5,6,7,8,9,10,11,12,13,14,15);
}

static __device__ __forceinline__ v8f wmma_f16(v16h a, v16h b, v8f c) {
  // D = A(16x32 f16) * B(32x16 f16) + C(16x16 f32)
  return __builtin_amdgcn_wmma_f32_16x16x32_f16(false, a, false, b, (short)0, c,
                                                false, false);
}

// =====================================================================
// Kernel 1: fused QKV projection.
//   out = (X @ W + b) -> f16   (Q additionally scaled by 1/sqrt(64))
// Grid: x = M/128 (64), y = 3*768/64 (36).  Block: 256 (8 waves).
// Each wave computes a 16x64 slab of the 128x64 output tile.
// =====================================================================
__global__ __launch_bounds__(256) void qkv_proj_kernel(
    const float* __restrict__ X,
    const float* __restrict__ Wq, const float* __restrict__ bq,
    const float* __restrict__ Wk, const float* __restrict__ bk,
    const float* __restrict__ Wv, const float* __restrict__ bv,
    _Float16* __restrict__ Qh, _Float16* __restrict__ Kh,
    _Float16* __restrict__ Vh) {
  const int tid  = threadIdx.x;
  const int wave = tid >> 5;
  const int lane = tid & 31;
  const int l16  = lane & 15;
  const int hi   = (lane >> 4) & 1;

  const int m0  = blockIdx.x * 128;
  const int mat = blockIdx.y / 12;        // 0=Q, 1=K, 2=V
  const int n0  = (blockIdx.y % 12) * 64; // column tile within [0,768)

  const float* __restrict__ W    = (mat == 0) ? Wq : ((mat == 1) ? Wk : Wv);
  const float* __restrict__ bias = (mat == 0) ? bq : ((mat == 1) ? bk : bv);
  _Float16*   __restrict__ Oh    = (mat == 0) ? Qh : ((mat == 1) ? Kh : Vh);
  const float oscale = (mat == 0) ? 0.125f : 1.0f;  // fold 1/sqrt(HEAD_DIM) into Q

  __shared__ _Float16 Xs[128][32];  // X tile, row-major (M x K)
  __shared__ _Float16 Ws[64][32];   // W tile, TRANSPOSED (N x K) -> contiguous B frags

  v8f acc[4] = {};

  for (int kk = 0; kk < HID / 32; ++kk) {
    // --- stage X tile (128x32 f32 -> f16) ---
#pragma unroll
    for (int i = 0; i < 4; ++i) {
      const int t   = tid + i * 256;      // 0..1023
      const int row = t >> 3;
      const int seg = t & 7;
      const float4 xv = *(const float4*)(X + (size_t)(m0 + row) * HID + kk * 32 + seg * 4);
      v4h hv;
      hv[0] = (_Float16)xv.x; hv[1] = (_Float16)xv.y;
      hv[2] = (_Float16)xv.z; hv[3] = (_Float16)xv.w;
      *(v4h*)(&Xs[row][seg * 4]) = hv;
    }
    // --- stage W tile transposed (32x64 f32 -> 64x32 f16) ---
#pragma unroll
    for (int i = 0; i < 2; ++i) {
      const int t  = tid + i * 256;       // 0..511
      const int n  = t & 63;
      const int kg = t >> 6;              // 0..7
      v4h wv;
#pragma unroll
      for (int j = 0; j < 4; ++j)
        wv[j] = (_Float16)W[(size_t)(kk * 32 + kg * 4 + j) * HID + n0 + n];
      *(v4h*)(&Ws[n][kg * 4]) = wv;
    }
    __syncthreads();

    // --- A fragment: rows wave*16..+15, K chunk of 32 ---
    const int arow = wave * 16 + l16;
    const int ab   = hi ? 16 : 0;
    const v8h alo  = *(const v8h*)(&Xs[arow][ab]);
    const v8h ahi  = *(const v8h*)(&Xs[arow][8 + ab]);
    const v16h a   = cat16(alo, ahi);

#pragma unroll
    for (int nb = 0; nb < 4; ++nb) {
      const int n  = nb * 16 + l16;
      const int kb = hi ? 16 : 0;
      const v16h b = *(const v16h*)(&Ws[n][kb]);
      acc[nb] = wmma_f16(a, b, acc[nb]);
    }
    __syncthreads();
  }

  // --- epilogue: bias add, optional Q-scale, f16 store ---
#pragma unroll
  for (int nb = 0; nb < 4; ++nb) {
    const int ncol = n0 + nb * 16 + l16;
    const float bv_ = bias[ncol];
#pragma unroll
    for (int r = 0; r < 8; ++r) {
      const int m = m0 + wave * 16 + r + hi * 8;
      const float val = (acc[nb][r] + bv_) * oscale;
      Oh[(size_t)m * HID + ncol] = (_Float16)val;
    }
  }
}

// =====================================================================
// Kernel 2: flash attention.
// Grid: x = B*NHEAD (48), y = SEQ/128 (16).  Block: 256 (8 waves).
// Each wave owns 16 query rows; 64-key tiles streamed through LDS.
// =====================================================================
__global__ __launch_bounds__(256) void flash_attn_kernel(
    const _Float16* __restrict__ Qh, const _Float16* __restrict__ Kh,
    const _Float16* __restrict__ Vh, const float* __restrict__ mask,
    float* __restrict__ out) {
  const int tid  = threadIdx.x;
  const int wave = tid >> 5;
  const int lane = tid & 31;
  const int l16  = lane & 15;
  const int hi   = (lane >> 4) & 1;

  const int bh = blockIdx.x;
  const int b  = bh / NHEAD;
  const int h  = bh % NHEAD;
  const int q0 = blockIdx.y * 128;

  __shared__ _Float16 Ks[64][64];        // K tile, row-major (key x d)
  __shared__ _Float16 Vts[64][64];       // V tile, TRANSPOSED (d x key)
  __shared__ _Float16 Ps[8][16][64];     // per-wave probs (row x key), C->A relayout

  // --- load this wave's Q A-fragments (16 rows x d=64 -> two 16x32 frags) ---
  const int qrow = q0 + wave * 16 + l16;
  const _Float16* qp = Qh + (size_t)(b * SEQ + qrow) * HID + h * HDIM;
  v16h qa[2];
#pragma unroll
  for (int j = 0; j < 2; ++j) {
    const int ab = j * 32 + (hi ? 16 : 0);
    const v8h lo = *(const v8h*)(qp + ab);
    const v8h hh = *(const v8h*)(qp + ab + 8);
    qa[j] = cat16(lo, hh);
  }

  float mrow[8], lrow[8];
#pragma unroll
  for (int r = 0; r < 8; ++r) { mrow[r] = -1e30f; lrow[r] = 0.0f; }
  v8f o[4] = {};

  for (int kt = 0; kt < SEQ / 64; ++kt) {
    const int key0 = kt * 64;
    // --- stage K tile + transposed V tile ---
#pragma unroll
    for (int i = 0; i < 2; ++i) {
      const int t   = tid + i * 256;     // 0..511
      const int row = t >> 3;
      const int cg  = t & 7;
      const size_t gro = (size_t)(b * SEQ + key0 + row) * HID + h * HDIM + cg * 8;
      const v8h kv = *(const v8h*)(Kh + gro);
      *(v8h*)(&Ks[row][cg * 8]) = kv;
      const v8h vv = *(const v8h*)(Vh + gro);
#pragma unroll
      for (int j = 0; j < 8; ++j) Vts[cg * 8 + j][row] = vv[j];
    }
    if (kt + 1 < SEQ / 64) {  // gfx1250 global_prefetch_b8 of next tile
      const size_t nxt = (size_t)(b * SEQ + key0 + 64 + (tid >> 2)) * HID + h * HDIM;
      __builtin_prefetch(Kh + nxt, 0, 1);
      __builtin_prefetch(Vh + nxt, 0, 1);
    }
    __syncthreads();

    // --- scores: S = Q * K^T  (4 blocks of 16 keys, 2 K-chunks each) ---
    v8f c[4];
#pragma unroll
    for (int blk = 0; blk < 4; ++blk) {
      v8f sacc = {};
#pragma unroll
      for (int j = 0; j < 2; ++j) {
        const int kb = hi ? 16 : 0;
        const v16h bf = *(const v16h*)(&Ks[blk * 16 + l16][j * 32 + kb]);
        sacc = wmma_f16(qa[j], bf, sacc);
      }
      const float mk = mask[b * SEQ + key0 + blk * 16 + l16];
#pragma unroll
      for (int r = 0; r < 8; ++r) sacc[r] += mk;
      c[blk] = sacc;
    }

    // --- online softmax: tile row-max, rescale O and l ---
    float mn[8];
#pragma unroll
    for (int r = 0; r < 8; ++r) {
      float tm = fmaxf(fmaxf(c[0][r], c[1][r]), fmaxf(c[2][r], c[3][r]));
#pragma unroll
      for (int off = 8; off >= 1; off >>= 1)
        tm = fmaxf(tm, __shfl_xor(tm, off, 16));
      const float mnew  = fmaxf(mrow[r], tm);
      const float alpha = __expf(mrow[r] - mnew);
      lrow[r] *= alpha;
      mrow[r] = mnew;
      mn[r]   = mnew;
#pragma unroll
      for (int nb = 0; nb < 4; ++nb) o[nb][r] *= alpha;
    }

    // --- P = exp(S - m), row-sums, C-layout -> A-layout via LDS ---
    float rs[8] = {0, 0, 0, 0, 0, 0, 0, 0};
#pragma unroll
    for (int blk = 0; blk < 4; ++blk) {
#pragma unroll
      for (int r = 0; r < 8; ++r) {
        const float p = __expf(c[blk][r] - mn[r]);
        rs[r] += p;
        Ps[wave][r + hi * 8][blk * 16 + l16] = (_Float16)p;
      }
    }
#pragma unroll
    for (int r = 0; r < 8; ++r) {
#pragma unroll
      for (int off = 8; off >= 1; off >>= 1)
        rs[r] += __shfl_xor(rs[r], off, 16);
      lrow[r] += rs[r];
    }
    __syncthreads();

    // --- O += P * V  (2 key-chunks of 32, 4 d-blocks of 16) ---
#pragma unroll
    for (int kc = 0; kc < 2; ++kc) {
      const int ab = hi ? 16 : 0;
      const v8h plo = *(const v8h*)(&Ps[wave][l16][kc * 32 + ab]);
      const v8h phi = *(const v8h*)(&Ps[wave][l16][kc * 32 + 8 + ab]);
      const v16h pa = cat16(plo, phi);
#pragma unroll
      for (int nb = 0; nb < 4; ++nb) {
        const int kb = hi ? 16 : 0;
        const v16h bf = *(const v16h*)(&Vts[nb * 16 + l16][kc * 32 + kb]);
        o[nb] = wmma_f16(pa, bf, o[nb]);
      }
    }
    __syncthreads();
  }

  // --- epilogue: normalize and write fp32 context [B,S,768] ---
#pragma unroll
  for (int nb = 0; nb < 4; ++nb) {
    const int col = h * HDIM + nb * 16 + l16;
#pragma unroll
    for (int r = 0; r < 8; ++r) {
      const int qr = q0 + wave * 16 + r + hi * 8;
      out[(size_t)(b * SEQ + qr) * HID + col] = o[nb][r] / lrow[r];
    }
  }
}

// =====================================================================
// Launcher
// =====================================================================
extern "C" void kernel_launch(void* const* d_in, const int* in_sizes, int n_in,
                              void* d_out, int out_size, void* d_ws, size_t ws_size,
                              hipStream_t stream) {
  const float* X    = (const float*)d_in[0];
  const float* mask = (const float*)d_in[1];
  const float* Wq   = (const float*)d_in[2];
  const float* bq   = (const float*)d_in[3];
  const float* Wk   = (const float*)d_in[4];
  const float* bk   = (const float*)d_in[5];
  const float* Wv   = (const float*)d_in[6];
  const float* bv   = (const float*)d_in[7];
  float* out        = (float*)d_out;

  // f16 Q/K/V staging in workspace: 3 * 8192*768*2 bytes = ~36 MB
  _Float16* Qh = (_Float16*)d_ws;
  _Float16* Kh = Qh + (size_t)NROW * HID;
  _Float16* Vh = Kh + (size_t)NROW * HID;

  dim3 g1(NROW / 128, 36);  // 64 x 36 tiles, 3 matrices fused along y
  qkv_proj_kernel<<<g1, 256, 0, stream>>>(X, Wq, bq, Wk, bk, Wv, bv, Qh, Kh, Vh);

  dim3 g2(BATCH * NHEAD, SEQ / 128);  // 48 x 16
  flash_attn_kernel<<<g2, 256, 0, stream>>>(Qh, Kh, Vh, mask, out);
}
